// DGCNN_64321430225655
// MI455X (gfx1250) — matrix operand order repaired and checked by reading, not compiled, not run on hardware
//
#include <hip/hip_runtime.h>
#include <hip/hip_bf16.h>
#include <cstdint>

#define BB  8
#define NN  2048
#define KNB 20
#define FF  6

typedef __attribute__((ext_vector_type(16))) _Float16 v16h;
typedef __attribute__((ext_vector_type(8)))  _Float16 v8h;
typedef __attribute__((ext_vector_type(8)))  float    v8f;

static __device__ inline v8f wmma_f16f32(v16h a, v16h b, v8f c) {
#if defined(__HIP_DEVICE_COMPILE__)
  return __builtin_amdgcn_wmma_f32_16x16x32_f16(false, a, false, b, (short)0, c, false, false);
#else
  return c;
#endif
}

// ---------------- weight prep: fp32 (cin,cout) -> f16 transposed+padded (coutp,cinp) ----
__global__ void prep_wt_kernel(const float* __restrict__ w, _Float16* __restrict__ wt,
                               int cin, int cout, int cinp, int coutp) {
  int g = blockIdx.x * 256 + threadIdx.x;
  if (g >= coutp * cinp) return;
  int o = g / cinp, k = g - o * cinp;
  float v = (o < cout && k < cin) ? w[(size_t)k * cout + o] : 0.0f;
  wt[g] = (_Float16)v;
}

// edge-conv first layer: w is (2C,cout); produce Wt of (Wc - Wn) and Wt of Wn
__global__ void prep_wt_split_kernel(const float* __restrict__ w, _Float16* __restrict__ wtD,
                                     _Float16* __restrict__ wtN, int C, int cout,
                                     int cinp, int coutp) {
  int g = blockIdx.x * 256 + threadIdx.x;
  if (g >= coutp * cinp) return;
  int o = g / cinp, k = g - o * cinp;
  float dv = 0.0f, nv = 0.0f;
  if (o < cout && k < C) {
    float wc = w[(size_t)k * cout + o];
    float wn = w[(size_t)(C + k) * cout + o];
    dv = wc - wn; nv = wn;
  }
  wtD[g] = (_Float16)dv;
  wtN[g] = (_Float16)nv;
}

// ---------------- KNN: top-20 smallest squared distances (includes self) ----------------
template <int CD>
__global__ void __launch_bounds__(128) knn_kernel(const float* __restrict__ X,
                                                  int* __restrict__ outIdx) {
  __shared__ float tile[128 * CD];
  const int b = blockIdx.y;
  const int i = blockIdx.x * 128 + threadIdx.x;
  const float* xb = X + (size_t)b * NN * CD;
  float xf[CD];
#pragma unroll
  for (int c = 0; c < CD; ++c) xf[c] = xb[(size_t)i * CD + c];
  float bd[KNB]; int bi[KNB];
#pragma unroll
  for (int k = 0; k < KNB; ++k) { bd[k] = 3.0e38f; bi[k] = 0; }
  for (int j0 = 0; j0 < NN; j0 += 128) {
    __syncthreads();
    for (int t = threadIdx.x; t < 128 * CD; t += 128) tile[t] = xb[(size_t)j0 * CD + t];
    __syncthreads();
    for (int jj = 0; jj < 128; ++jj) {
      float d = 0.0f;
#pragma unroll
      for (int c = 0; c < CD; ++c) { float df = xf[c] - tile[jj * CD + c]; d = fmaf(df, df, d); }
      if (d < bd[KNB - 1]) {           // bubble-insert, constant indices (stays in VGPRs)
        float cd = d; int ci = j0 + jj;
#pragma unroll
        for (int q = 0; q < KNB; ++q) {
          if (cd < bd[q]) { float td = bd[q]; int ti = bi[q];
                            bd[q] = cd; bi[q] = ci; cd = td; ci = ti; }
        }
      }
    }
  }
#pragma unroll
  for (int k = 0; k < KNB; ++k) outIdx[((size_t)b * NN + i) * KNB + k] = bi[k];
}

// ---------------- generic WMMA GEMM: out = act((X @ W + b) * s + t) ---------------------
// X fp32 (rows,cin); Wt f16 (coutp,cinp) = W^T zero-padded. block=128 (4 waves),
// grid=(rows/16, coutp/64). maxmode: atomicMax of relu'd bits into out[(row/NN)*cout+col].
__global__ void __launch_bounds__(128)
gemm_kernel(const float* __restrict__ X, const _Float16* __restrict__ Wt,
            const float* __restrict__ bias, const float* __restrict__ sc,
            const float* __restrict__ sh, float* __restrict__ out,
            int cin, int cinp, int cout, int relu, int maxmode, int rowsPerBatch) {
  extern __shared__ __align__(16) _Float16 sA[];  // 16 x cinp halves
  const int tid = threadIdx.x;
  const int row0 = blockIdx.x * 16;
  for (int i = tid; i < 16 * cinp; i += 128) {
    int r = i / cinp, c = i % cinp;
    float v = (c < cin) ? X[(size_t)(row0 + r) * cin + c] : 0.0f;
    sA[i] = (_Float16)v;
  }
  __syncthreads();
  const int wv = tid >> 5, lane = tid & 31;
  const int rl = lane & 15, hi = lane >> 4;
  const int c0 = blockIdx.y * 64 + wv * 16;
  v8f acc = {};
  for (int k0 = 0; k0 < cinp; k0 += 32) {
    const _Float16* pa = sA + rl * cinp + k0 + hi * 8;           // ISA A layout (16x32 f16)
    v8h a0 = *(const v8h*)pa;
    v8h a1 = *(const v8h*)(pa + 16);
    const _Float16* pb = Wt + (size_t)(c0 + rl) * cinp + k0 + hi * 8;  // ISA B layout
    v8h b0 = *(const v8h*)pb;
    v8h b1 = *(const v8h*)(pb + 16);
    v16h a, b;
#pragma unroll
    for (int t = 0; t < 8; ++t) { a[t] = a0[t]; a[t + 8] = a1[t];
                                  b[t] = b0[t]; b[t + 8] = b1[t]; }
    acc = wmma_f16f32(a, b, acc);
  }
  const int col = c0 + rl;
  if (col < cout) {
    float bb = bias ? bias[col] : 0.0f;
    float ss = sc ? sc[col] : 1.0f;
    float tt = sh ? sh[col] : 0.0f;
#pragma unroll
    for (int i = 0; i < 8; ++i) {
      int r = row0 + i + 8 * hi;                 // C/D layout: vgpr i -> row i + 8*hi
      float v = (acc[i] + bb) * ss + tt;
      if (relu) v = fmaxf(v, 0.0f);
      if (maxmode) {
        atomicMax((int*)out + ((size_t)(r / rowsPerBatch) * cout + col), __float_as_int(v));
      } else {
        out[(size_t)r * cout + col] = v;
      }
    }
  }
}

// ---------------- fused EdgeConv: gather h1 -> WMMA conv2 -> ReLU -> max over K ----------
// h1[edge] = relu((Xc[i] + Xn[j]) * s1 + t1), 64 ch. Then (320-edge x 64) @ W2t -> cout2,
// affine+relu, K-max via int-bit atomicMax in LDS. block=256 (8 waves), 16 points/block.
__global__ void __launch_bounds__(256)
edgeconv_kernel(const float* __restrict__ Xc, const float* __restrict__ Xn,
                const int* __restrict__ idx, const float* __restrict__ s1,
                const float* __restrict__ t1, const _Float16* __restrict__ W2t,
                const float* __restrict__ b2, const float* __restrict__ s2,
                const float* __restrict__ t2, float* __restrict__ out, int cout2) {
  __shared__ __align__(16) _Float16 h1s[320 * 64];  // 40 KB
  __shared__ int accS[16 * 128];                    // 8 KB (non-negative float bits)
  const int tid = threadIdx.x;
  const int b = blockIdx.y;
  const int pt0 = blockIdx.x * 16;

  for (int i = tid; i < 16 * cout2; i += 256) accS[i] = 0;
  for (int e = tid; e < 320 * 64; e += 256) {
    int edge = e >> 6, c = e & 63;
    int p = edge / 20, k = edge - p * 20;
    int i = pt0 + p;
    int j = idx[((size_t)b * NN + i) * KNB + k];
    float v = Xc[((size_t)b * NN + i) * 64 + c] + Xn[((size_t)b * NN + j) * 64 + c];
    h1s[edge * 64 + c] = (_Float16)fmaxf(v * s1[c] + t1[c], 0.0f);
  }
  __syncthreads();

  const int wv = tid >> 5, lane = tid & 31;
  const int rl = lane & 15, hi = lane >> 4;
  const int ntc = cout2 >> 4;
  for (int job = wv; job < 20 * ntc; job += 8) {   // wave-uniform: EXEC all-ones at WMMA
    int rt = job / ntc, ct = job - rt * ntc;
    v8f acc = {};
#pragma unroll
    for (int k0 = 0; k0 < 64; k0 += 32) {
      const _Float16* pa = h1s + (rt * 16 + rl) * 64 + k0 + hi * 8;
      v8h a0 = *(const v8h*)pa, a1 = *(const v8h*)(pa + 16);
      const _Float16* pb = W2t + (size_t)(ct * 16 + rl) * 64 + k0 + hi * 8;
      v8h b0 = *(const v8h*)pb, b1 = *(const v8h*)(pb + 16);
      v16h a, bm;
#pragma unroll
      for (int t = 0; t < 8; ++t) { a[t] = a0[t]; a[t + 8] = a1[t];
                                    bm[t] = b0[t]; bm[t + 8] = b1[t]; }
      acc = wmma_f16f32(a, bm, acc);
    }
    int col = ct * 16 + rl;
    float bb = b2[col], ss = s2[col], tt = t2[col];
#pragma unroll
    for (int i = 0; i < 8; ++i) {
      int r = rt * 16 + i + 8 * hi;
      int p = r / 20;
      float v = fmaxf((acc[i] + bb) * ss + tt, 0.0f);
      atomicMax(&accS[p * cout2 + col], __float_as_int(v));
    }
  }
  __syncthreads();
  for (int i = tid; i < 16 * cout2; i += 256) {
    int p = i / cout2, c = i - p * cout2;
    out[((size_t)b * NN + pt0 + p) * cout2 + c] = __int_as_float(accS[i]);
  }
}

// ---------------- EdgeConv3 (single conv): elementwise affine+relu, max over K ----------
__global__ void edge_max_kernel(const float* __restrict__ Xc, const float* __restrict__ Xn,
                                const int* __restrict__ idx, const float* __restrict__ s,
                                const float* __restrict__ t, float* __restrict__ out) {
  int g = blockIdx.x * 256 + threadIdx.x;  // B*N*64
  int c = g & 63;
  int pi = g >> 6;
  int b = pi >> 11;
  float xc = Xc[g];
  float ss = s[c], tt = t[c];
  float m = -3.0e38f;
  const int* ip = idx + (size_t)pi * KNB;
  for (int k = 0; k < KNB; ++k) {
    int j = ip[k];
    float v = fmaxf((xc + Xn[(((size_t)b << 11) + j) * 64 + c]) * ss + tt, 0.0f);
    m = fmaxf(m, v);
  }
  out[g] = m;
}

// ---------------- transform-net tail: tD -> tE -> T = h@tw + tb (8 rows, tiny) ----------
__global__ void tnet_head_kernel(const float* __restrict__ gt,
    const float* __restrict__ tDw, const float* __restrict__ tDb,
    const float* __restrict__ tDs, const float* __restrict__ tDt,
    const float* __restrict__ tEw, const float* __restrict__ tEb,
    const float* __restrict__ tEs, const float* __restrict__ tEt,
    const float* __restrict__ tw, const float* __restrict__ tb, float* __restrict__ T) {
  __shared__ float s_in[1024]; __shared__ float s_h[512]; __shared__ float s_h2[256];
  int b = blockIdx.x, t = threadIdx.x;
  for (int c = t; c < 1024; c += 256) s_in[c] = gt[b * 1024 + c];
  __syncthreads();
  for (int o = t; o < 512; o += 256) {
    float a = tDb[o];
    for (int k = 0; k < 1024; ++k) a = fmaf(s_in[k], tDw[(size_t)k * 512 + o], a);
    s_h[o] = fmaxf(a * tDs[o] + tDt[o], 0.0f);
  }
  __syncthreads();
  for (int o = t; o < 256; o += 256) {
    float a = tEb[o];
    for (int k = 0; k < 512; ++k) a = fmaf(s_h[k], tEw[(size_t)k * 256 + o], a);
    s_h2[o] = fmaxf(a * tEs[o] + tEt[o], 0.0f);
  }
  __syncthreads();
  for (int o = t; o < FF * FF; o += 256) {
    float a = tb[o];
    for (int k = 0; k < 256; ++k) a = fmaf(s_h2[k], tw[(size_t)k * (FF * FF) + o], a);
    T[b * (FF * FF) + o] = a;
  }
}

__global__ void apply_T_kernel(const float* __restrict__ pc, const float* __restrict__ T,
                               float* __restrict__ pct) {
  int g = blockIdx.x * 256 + threadIdx.x;  // B*N
  int b = g >> 11;
  float x[FF];
#pragma unroll
  for (int c = 0; c < FF; ++c) x[c] = pc[(size_t)g * FF + c];
#pragma unroll
  for (int f = 0; f < FF; ++f) {
    float a = 0.0f;
#pragma unroll
    for (int c = 0; c < FF; ++c) a = fmaf(x[c], T[b * 36 + c * FF + f], a);
    pct[(size_t)g * FF + f] = a;
  }
}

__global__ void concat3_kernel(const float* __restrict__ a, const float* __restrict__ bsrc,
                               const float* __restrict__ csrc, float* __restrict__ out) {
  int g = blockIdx.x * 256 + threadIdx.x;  // B*N*192
  int ch = g % 192; int p = g / 192;
  float v;
  if (ch < 64)       v = a[(size_t)p * 64 + ch];
  else if (ch < 128) v = bsrc[(size_t)p * 64 + ch - 64];
  else               v = csrc[(size_t)p * 64 + ch - 128];
  out[g] = v;
}

// plain fp32 matmul (+optional bias) for head-weight collapse: C = A@B (+bias per col)
__global__ void matmul_plain_kernel(const float* __restrict__ A, const float* __restrict__ Bm,
                                    const float* __restrict__ bias, float* __restrict__ Cm,
                                    int M, int Kd, int Nd) {
  int g = blockIdx.x * 256 + threadIdx.x;
  if (g >= M * Nd) return;
  int m = g / Nd, n = g - m * Nd;
  float a = bias ? bias[n] : 0.0f;
  for (int k = 0; k < Kd; ++k) a = fmaf(A[(size_t)m * Kd + k], Bm[(size_t)k * Nd + n], a);
  Cm[g] = a;
}

// fused head: out = [g, net1, net2, net3] @ Wtot + btot  (1216 -> 8, Wtot cached in LDS)
__global__ void __launch_bounds__(256)
head_kernel(const float* __restrict__ gf, const float* __restrict__ n1,
            const float* __restrict__ n2, const float* __restrict__ n3,
            const float* __restrict__ Wtot, const float* __restrict__ btot,
            float* __restrict__ out) {
  __shared__ float sW[1216 * 8];
  __shared__ float sg[1024];
  int b = blockIdx.y, t = threadIdx.x;
  for (int i = t; i < 1216 * 8; i += 256) sW[i] = Wtot[i];
  for (int i = t; i < 1024; i += 256) sg[i] = gf[b * 1024 + i];
  __syncthreads();
  int p = blockIdx.x * 256 + t;
  size_t pi = (size_t)b * NN + p;
  float acc[8];
#pragma unroll
  for (int c = 0; c < 8; ++c) acc[c] = btot[c];
  for (int k = 0; k < 1024; ++k) {
    float v = sg[k];
#pragma unroll
    for (int c = 0; c < 8; ++c) acc[c] = fmaf(v, sW[k * 8 + c], acc[c]);
  }
  const float* r1 = n1 + pi * 64;
  for (int k = 0; k < 64; ++k) { float v = r1[k];
#pragma unroll
    for (int c = 0; c < 8; ++c) acc[c] = fmaf(v, sW[(1024 + k) * 8 + c], acc[c]); }
  const float* r2 = n2 + pi * 64;
  for (int k = 0; k < 64; ++k) { float v = r2[k];
#pragma unroll
    for (int c = 0; c < 8; ++c) acc[c] = fmaf(v, sW[(1088 + k) * 8 + c], acc[c]); }
  const float* r3 = n3 + pi * 64;
  for (int k = 0; k < 64; ++k) { float v = r3[k];
#pragma unroll
    for (int c = 0; c < 8; ++c) acc[c] = fmaf(v, sW[(1152 + k) * 8 + c], acc[c]); }
#pragma unroll
  for (int c = 0; c < 8; ++c) out[pi * 8 + c] = acc[c];
}

// =========================================================================================
extern "C" void kernel_launch(void* const* d_in, const int* in_sizes, int n_in,
                              void* d_out, int out_size, void* d_ws, size_t ws_size,
                              hipStream_t stream) {
  (void)in_sizes; (void)n_in; (void)out_size; (void)ws_size;
  // setup_inputs() flatten order: point_cloud, then params dict (each conv: w,b[,s,t])
  const float* pc  = (const float*)d_in[0];
  const float* tAw = (const float*)d_in[1],  *tAb = (const float*)d_in[2];
  const float* tAs = (const float*)d_in[3],  *tAt = (const float*)d_in[4];
  const float* tBw = (const float*)d_in[5],  *tBb = (const float*)d_in[6];
  const float* tBs = (const float*)d_in[7],  *tBt = (const float*)d_in[8];
  const float* tCw = (const float*)d_in[9],  *tCb = (const float*)d_in[10];
  const float* tCs = (const float*)d_in[11], *tCt = (const float*)d_in[12];
  const float* tDw = (const float*)d_in[13], *tDb = (const float*)d_in[14];
  const float* tDs = (const float*)d_in[15], *tDt = (const float*)d_in[16];
  const float* tEw = (const float*)d_in[17], *tEb = (const float*)d_in[18];
  const float* tEs = (const float*)d_in[19], *tEt = (const float*)d_in[20];
  const float* tw  = (const float*)d_in[21], *tb  = (const float*)d_in[22];
  const float* Aw  = (const float*)d_in[23], *Ab  = (const float*)d_in[24];
  const float* As  = (const float*)d_in[25], *At  = (const float*)d_in[26];
  const float* Bcw = (const float*)d_in[27], *Bcb = (const float*)d_in[28];
  const float* Bcs = (const float*)d_in[29], *Bct = (const float*)d_in[30];
  const float* Ccw = (const float*)d_in[31], *Ccb = (const float*)d_in[32];
  const float* Ccs = (const float*)d_in[33], *Cct = (const float*)d_in[34];
  const float* Dw  = (const float*)d_in[35], *Db  = (const float*)d_in[36];
  const float* Ds  = (const float*)d_in[37], *Dt  = (const float*)d_in[38];
  const float* Ew  = (const float*)d_in[39], *Eb  = (const float*)d_in[40];
  const float* Es  = (const float*)d_in[41], *Et  = (const float*)d_in[42];
  const float* Fcw = (const float*)d_in[43], *Fcb = (const float*)d_in[44];
  const float* Fcs = (const float*)d_in[45], *Fct = (const float*)d_in[46];
  const float* Gw  = (const float*)d_in[47], *Gb  = (const float*)d_in[48];
  const float* Hw  = (const float*)d_in[49], *Hb  = (const float*)d_in[50];
  const float* Iw  = (const float*)d_in[51], *Ib  = (const float*)d_in[52];
  const float* Jw  = (const float*)d_in[53], *Jb  = (const float*)d_in[54];

  // ---- workspace carving (deterministic) ----
  char* wp = (char*)d_ws;
  auto alloc = [&](size_t bytes) -> void* {
    void* p = (void*)wp; wp += (bytes + 255) & ~(size_t)255; return p;
  };
  _Float16* wtAD = (_Float16*)alloc(64 * 32 * 2);
  _Float16* wtAN = (_Float16*)alloc(64 * 32 * 2);
  _Float16* wtB  = (_Float16*)alloc(128 * 64 * 2);
  _Float16* wtC  = (_Float16*)alloc(1024 * 128 * 2);
  _Float16* wAD  = (_Float16*)alloc(64 * 32 * 2);
  _Float16* wAN  = (_Float16*)alloc(64 * 32 * 2);
  _Float16* wBc  = (_Float16*)alloc(64 * 64 * 2);
  _Float16* wCcD = (_Float16*)alloc(64 * 64 * 2);
  _Float16* wCcN = (_Float16*)alloc(64 * 64 * 2);
  _Float16* wDw  = (_Float16*)alloc(64 * 64 * 2);
  _Float16* wED  = (_Float16*)alloc(64 * 64 * 2);
  _Float16* wEN  = (_Float16*)alloc(64 * 64 * 2);
  _Float16* wFc  = (_Float16*)alloc(1024 * 192 * 2);
  float* XC    = (float*)alloc((size_t)BB * NN * 64 * 4);
  float* XN    = (float*)alloc((size_t)BB * NN * 64 * 4);
  float* net1  = (float*)alloc((size_t)BB * NN * 64 * 4);
  float* net2  = (float*)alloc((size_t)BB * NN * 64 * 4);
  float* net3  = (float*)alloc((size_t)BB * NN * 64 * 4);
  float* h128  = (float*)alloc((size_t)BB * NN * 128 * 4);
  float* pct   = (float*)alloc((size_t)BB * NN * FF * 4);
  int*   idxB  = (int*)alloc((size_t)BB * NN * KNB * 4);
  float* gt    = (float*)alloc(BB * 1024 * 4);
  float* gf    = (float*)alloc(BB * 1024 * 4);
  float* Tm    = (float*)alloc(BB * 36 * 4);
  float* combo = (float*)alloc((size_t)BB * NN * 192 * 4);
  float* W12   = (float*)alloc(1216 * 256 * 4);
  float* W123  = (float*)alloc(1216 * 128 * 4);
  float* Wtot  = (float*)alloc(1216 * 8 * 4);
  float* b12   = (float*)alloc(256 * 4);
  float* b123  = (float*)alloc(128 * 4);
  float* btot  = (float*)alloc(8 * 4);

  dim3 blk256(256), blk128(128);
  auto g1 = [](int n) { return dim3((unsigned)((n + 255) / 256)); };
  const int ROWS = BB * NN;           // 16384
  dim3 gemmRows(ROWS / 16, 1);

  // ---- weight prep ----
  prep_wt_split_kernel<<<g1(64 * 32), blk256, 0, stream>>>(tAw, wtAD, wtAN, FF, 64, 32, 64);
  prep_wt_kernel<<<g1(128 * 64), blk256, 0, stream>>>(tBw, wtB, 64, 128, 64, 128);
  prep_wt_kernel<<<g1(1024 * 128), blk256, 0, stream>>>(tCw, wtC, 128, 1024, 128, 1024);
  prep_wt_split_kernel<<<g1(64 * 32), blk256, 0, stream>>>(Aw, wAD, wAN, FF, 64, 32, 64);
  prep_wt_kernel<<<g1(64 * 64), blk256, 0, stream>>>(Bcw, wBc, 64, 64, 64, 64);
  prep_wt_split_kernel<<<g1(64 * 64), blk256, 0, stream>>>(Ccw, wCcD, wCcN, 64, 64, 64, 64);
  prep_wt_kernel<<<g1(64 * 64), blk256, 0, stream>>>(Dw, wDw, 64, 64, 64, 64);
  prep_wt_split_kernel<<<g1(64 * 64), blk256, 0, stream>>>(Ew, wED, wEN, 64, 64, 64, 64);
  prep_wt_kernel<<<g1(1024 * 192), blk256, 0, stream>>>(Fcw, wFc, 192, 1024, 192, 1024);

  // ---- collapse linear head: Wtot = Gw@Hw@Iw@Jw, btot = ((Gb@Hw+Hb)@Iw+Ib)@Jw+Jb ----
  matmul_plain_kernel<<<g1(1216 * 256), blk256, 0, stream>>>(Gw, Hw, nullptr, W12, 1216, 256, 256);
  matmul_plain_kernel<<<g1(1216 * 128), blk256, 0, stream>>>(W12, Iw, nullptr, W123, 1216, 256, 128);
  matmul_plain_kernel<<<g1(1216 * 8), blk256, 0, stream>>>(W123, Jw, nullptr, Wtot, 1216, 128, 8);
  matmul_plain_kernel<<<g1(256), blk256, 0, stream>>>(Gb, Hw, Hb, b12, 1, 256, 256);
  matmul_plain_kernel<<<g1(128), blk256, 0, stream>>>(b12, Iw, Ib, b123, 1, 256, 128);
  matmul_plain_kernel<<<g1(8), blk256, 0, stream>>>(b123, Jw, Jb, btot, 1, 128, 8);

  // ---- input transform net ----
  knn_kernel<FF><<<dim3(NN / 128, BB), blk128, 0, stream>>>(pc, idxB);
  gemm_kernel<<<gemmRows, blk128, 16 * 32 * 2, stream>>>(pc, wtAD, tAb, nullptr, nullptr, XC, FF, 32, 64, 0, 0, NN);
  gemm_kernel<<<gemmRows, blk128, 16 * 32 * 2, stream>>>(pc, wtAN, nullptr, nullptr, nullptr, XN, FF, 32, 64, 0, 0, NN);
  edgeconv_kernel<<<dim3(NN / 16, BB), blk256, 0, stream>>>(XC, XN, idxB, tAs, tAt, wtB, tBb, tBs, tBt, h128, 128);
  hipMemsetAsync(gt, 0, BB * 1024 * 4, stream);
  gemm_kernel<<<dim3(ROWS / 16, 16), blk128, 16 * 128 * 2, stream>>>(h128, wtC, tCb, tCs, tCt, gt, 128, 128, 1024, 1, 1, NN);
  tnet_head_kernel<<<dim3(BB), blk256, 0, stream>>>(gt, tDw, tDb, tDs, tDt, tEw, tEb, tEs, tEt, tw, tb, Tm);
  apply_T_kernel<<<g1(ROWS), blk256, 0, stream>>>(pc, Tm, pct);

  // ---- EdgeConv 1 ----
  knn_kernel<FF><<<dim3(NN / 128, BB), blk128, 0, stream>>>(pct, idxB);
  gemm_kernel<<<gemmRows, blk128, 16 * 32 * 2, stream>>>(pct, wAD, Ab, nullptr, nullptr, XC, FF, 32, 64, 0, 0, NN);
  gemm_kernel<<<gemmRows, blk128, 16 * 32 * 2, stream>>>(pct, wAN, nullptr, nullptr, nullptr, XN, FF, 32, 64, 0, 0, NN);
  edgeconv_kernel<<<dim3(NN / 16, BB), blk256, 0, stream>>>(XC, XN, idxB, As, At, wBc, Bcb, Bcs, Bct, net1, 64);

  // ---- EdgeConv 2 (dynamic graph on net1) ----
  knn_kernel<64><<<dim3(NN / 128, BB), blk128, 0, stream>>>(net1, idxB);
  gemm_kernel<<<gemmRows, blk128, 16 * 64 * 2, stream>>>(net1, wCcD, Ccb, nullptr, nullptr, XC, 64, 64, 64, 0, 0, NN);
  gemm_kernel<<<gemmRows, blk128, 16 * 64 * 2, stream>>>(net1, wCcN, nullptr, nullptr, nullptr, XN, 64, 64, 64, 0, 0, NN);
  edgeconv_kernel<<<dim3(NN / 16, BB), blk256, 0, stream>>>(XC, XN, idxB, Ccs, Cct, wDw, Db, Ds, Dt, net2, 64);

  // ---- EdgeConv 3 (single conv) ----
  knn_kernel<64><<<dim3(NN / 128, BB), blk128, 0, stream>>>(net2, idxB);
  gemm_kernel<<<gemmRows, blk128, 16 * 64 * 2, stream>>>(net2, wED, Eb, nullptr, nullptr, XC, 64, 64, 64, 0, 0, NN);
  gemm_kernel<<<gemmRows, blk128, 16 * 64 * 2, stream>>>(net2, wEN, nullptr, nullptr, nullptr, XN, 64, 64, 64, 0, 0, NN);
  edge_max_kernel<<<g1(ROWS * 64), blk256, 0, stream>>>(XC, XN, idxB, Es, Et, net3);

  // ---- Fc conv + fused global max over N ----
  concat3_kernel<<<g1(ROWS * 192), blk256, 0, stream>>>(net1, net2, net3, combo);
  hipMemsetAsync(gf, 0, BB * 1024 * 4, stream);
  gemm_kernel<<<dim3(ROWS / 16, 16), blk128, 16 * 192 * 2, stream>>>(combo, wFc, Fcb, Fcs, Fct, gf, 192, 192, 1024, 1, 1, NN);

  // ---- collapsed segmentation head ----
  head_kernel<<<dim3(NN / 256, BB), blk256, 0, stream>>>(gf, net1, net2, net3, Wtot, btot, (float*)d_out);
}
